// Mymodel_85117661872398
// MI455X (gfx1250) — matrix operand order repaired and checked by reading, not compiled, not run on hardware
//
#include <hip/hip_runtime.h>
#include <hip/hip_bf16.h>
#include <math.h>

// Problem dims (compile-time, from reference)
#define SS 128
#define BB 32
#define TT 64
#define HH 512
#define HD2 1024
#define VV 32000
#define KG 1536   // 1024 + 512 concatenated K for gates / logits GEMMs
#define NG 2048   // 4H gate width

typedef __bf16 bf16_t;
typedef __attribute__((ext_vector_type(16))) __bf16 v16bf;
typedef __attribute__((ext_vector_type(8)))  __bf16 v8bf;
typedef __attribute__((ext_vector_type(8)))  float  v8f;

// ---------------------------------------------------------------------------
// WMMA fragment layouts (CDNA5 wave32, CDNA5 ISA):
// A (16x32 bf16): lanes 0-15 = rows M0-15; half = lane>>4 selects K-groups:
//   elems 0-7 = K[half*8 .. +7], elems 8-15 = K[16+half*8 .. +7]
// B (32x16 bf16): lane&15 = column N; lanes0-15 hold K=0-7,16-23, lanes16-31 K=8-15,24-31
//   (two 8-K runs per lane, same order as A, from WT[n][k], k-contiguous)
// C/D (16x16 f32, 8 VGPRs): n = lane&15, rows (lane<16 ? r : 8+r)
//
// MT x NT register tiling: A fragments loaded once per k-step, reused over NT
// B tiles -> 2 loads/WMMA instead of 3; NT=2 keeps 1000 waves live on the
// dominant [32x1536]x[1536x32000] GEMM for L2-latency hiding.
// ---------------------------------------------------------------------------

template<int MT, int NT>
__global__ void gemm_wmma(const bf16_t* __restrict__ X, int lda,
                          const bf16_t* __restrict__ WT,
                          float* __restrict__ out, int ldo,
                          int m_groups, int n_groups, int K) {
  const int wave = (int)((blockIdx.x * blockDim.x + threadIdx.x) >> 5);
  if (wave >= m_groups * n_groups) return;
  const int mg   = wave / n_groups;
  const int ng   = wave % n_groups;
  const int lane = threadIdx.x & 31;
  const int half = lane >> 4;
  const int l16  = lane & 15;

  v8f acc[MT][NT];
#pragma unroll
  for (int i = 0; i < MT; ++i)
#pragma unroll
    for (int j = 0; j < NT; ++j)
      acc[i][j] = (v8f){0.f,0.f,0.f,0.f,0.f,0.f,0.f,0.f};

  const bf16_t* xrow = X + (size_t)(mg * MT * 16 + l16) * lda + half * 8;
  const bf16_t* wrow[NT];
#pragma unroll
  for (int j = 0; j < NT; ++j)
    wrow[j] = WT + (size_t)((ng * NT + j) * 16 + l16) * K + half * 8;

  for (int k0 = 0; k0 < K; k0 += 32) {
    v16bf af[MT];
#pragma unroll
    for (int i = 0; i < MT; ++i) {
      const bf16_t* xp = xrow + (size_t)i * 16 * lda + k0;
      v8bf a0 = *(const v8bf*)xp;          // K = k0 + half*8 .. +7
      v8bf a1 = *(const v8bf*)(xp + 16);   // K = k0 + 16 + half*8 .. +7
      af[i] = __builtin_shufflevector(a0, a1, 0,1,2,3,4,5,6,7,8,9,10,11,12,13,14,15);
    }
#pragma unroll
    for (int j = 0; j < NT; ++j) {
      v8bf b0 = *(const v8bf*)(wrow[j] + k0);
      v8bf b1 = *(const v8bf*)(wrow[j] + k0 + 16);
      v16bf bf = __builtin_shufflevector(b0, b1, 0,1,2,3,4,5,6,7,8,9,10,11,12,13,14,15);
#pragma unroll
      for (int i = 0; i < MT; ++i)
        acc[i][j] = __builtin_amdgcn_wmma_f32_16x16x32_bf16(
            false, af[i], false, bf, (short)0, acc[i][j], false, false);
    }
  }
#pragma unroll
  for (int i = 0; i < MT; ++i) {
    const int mbase = mg * MT * 16 + i * 16 + half * 8;
#pragma unroll
    for (int j = 0; j < NT; ++j) {
      const int nO = (ng * NT + j) * 16 + l16;
#pragma unroll
      for (int r = 0; r < 8; ++r)
        out[(size_t)(mbase + r) * ldo + nO] = acc[i][j][r];
    }
  }
}

// ---------------------------------------------------------------------------
// One-time conversion kernels
// ---------------------------------------------------------------------------

// dst[n * dst_ld + dst_off + k] = (bf16) src[k * cols + n]   (LDS-tiled transpose)
__global__ void transpose_cvt_kernel(const float* __restrict__ src, bf16_t* __restrict__ dst,
                                     int rows, int cols, int dst_ld, int dst_off) {
  __shared__ float tile[32][33];
  const int c0 = blockIdx.x * 32, r0 = blockIdx.y * 32;
  const int tx = threadIdx.x, ty = threadIdx.y;  // 32 x 8
#pragma unroll
  for (int i = 0; i < 32; i += 8) {
    int r = r0 + ty + i, c = c0 + tx;
    tile[ty + i][tx] = (r < rows && c < cols) ? src[(size_t)r * cols + c] : 0.f;
  }
  __syncthreads();
#pragma unroll
  for (int i = 0; i < 32; i += 8) {
    int n = c0 + ty + i, k = r0 + tx;
    if (n < cols && k < rows)
      dst[(size_t)n * dst_ld + dst_off + k] = (bf16_t)tile[tx][ty + i];
  }
}

// WgT[n*1536 + k] = W_ih[n][k] (k<1024) else W_hh[n][k-1024]   (fully coalesced)
__global__ void build_wgt_kernel(const float* __restrict__ W_ih, const float* __restrict__ W_hh,
                                 bf16_t* __restrict__ WgT) {
  size_t i = (size_t)blockIdx.x * blockDim.x + threadIdx.x;
  if (i >= (size_t)NG * KG) return;
  int n = (int)(i / KG), k = (int)(i % KG);
  float v = (k < HD2) ? W_ih[(size_t)n * HD2 + k] : W_hh[(size_t)n * HH + (k - HD2)];
  WgT[i] = (bf16_t)v;
}

__global__ void cvt_bf16_kernel(const float* __restrict__ src, bf16_t* __restrict__ dst, size_t n) {
  size_t i = (size_t)blockIdx.x * blockDim.x + threadIdx.x;
  if (i < n) dst[i] = (bf16_t)src[i];
}

__global__ void bias_kernel(const float* __restrict__ b_ih, const float* __restrict__ b_hh,
                            float* __restrict__ bias) {
  int i = blockIdx.x * blockDim.x + threadIdx.x;
  if (i < NG) bias[i] = b_ih[i] + b_hh[i];
}

// ---------------------------------------------------------------------------
// Per-step kernels
// ---------------------------------------------------------------------------

// scores[s*B+b] = sum_h tanh(wa_h[s*B+b][h] + sv[b][h]) * u_a[h]; one wave per (s,b)
__global__ void attn_scores_kernel(const float* __restrict__ wa_h, const float* __restrict__ sv,
                                   const float* __restrict__ u_a, float* __restrict__ scores) {
  const int wave = (int)((blockIdx.x * blockDim.x + threadIdx.x) >> 5);
  const int lane = threadIdx.x & 31;
  if (wave >= SS * BB) return;
  const int b = wave & (BB - 1);
  const float* wr = wa_h + (size_t)wave * HH;
  const float* sr = sv + (size_t)b * HH;
  float acc = 0.f;
  for (int i = lane; i < HH; i += 32) acc += tanhf(wr[i] + sr[i]) * u_a[i];
#pragma unroll
  for (int off = 16; off; off >>= 1) acc += __shfl_down(acc, off);
  if (lane == 0) scores[wave] = acc;
}

// softmax over S per batch column (scores * s_mask, axis 0)
__global__ void alpha_kernel(const float* __restrict__ scores, const float* __restrict__ s_mask,
                             float* __restrict__ alpha) {
  const int b = blockIdx.x;        // 32 blocks
  const int s = threadIdx.x;       // 128 threads
  __shared__ float red[SS];
  float v = scores[s * BB + b] * s_mask[s * BB + b];
  red[s] = v; __syncthreads();
  for (int off = 64; off; off >>= 1) { if (s < off) red[s] = fmaxf(red[s], red[s + off]); __syncthreads(); }
  float m = red[0]; __syncthreads();
  float e = expf(v - m);
  red[s] = e; __syncthreads();
  for (int off = 64; off; off >>= 1) { if (s < off) red[s] += red[s + off]; __syncthreads(); }
  alpha[s * BB + b] = e / red[0];
}

// ct[b][j] = sum_s alpha[s][b] * h_s[s][b][j]; writes bf16 into both GEMM inputs
__global__ void context_kernel(const float* __restrict__ alpha, const float* __restrict__ h_s,
                               bf16_t* __restrict__ xg, bf16_t* __restrict__ xo) {
  const int b = blockIdx.x;        // 32 blocks, 256 threads
  const int tid = threadIdx.x;
  __shared__ float al[SS];
  if (tid < SS) al[tid] = alpha[tid * BB + b];
  __syncthreads();
  for (int j = tid; j < HD2; j += 256) {
    float acc = 0.f;
#pragma unroll 4
    for (int s = 0; s < SS; ++s) acc += al[s] * h_s[((size_t)s * BB + b) * HD2 + j];
    bf16_t cb = (bf16_t)acc;
    xg[(size_t)b * KG + j] = cb;          // gates input: [ct | s]
    xo[(size_t)b * KG + HH + j] = cb;     // logits input: [s_new | ct]
  }
}

__device__ __forceinline__ float sigm(float x) { return 1.f / (1.f + expf(-x)); }

// LSTM cell elementwise: gates -> (c_new, s_new); s_new written bf16 into next inputs
__global__ void lstm_kernel(const float* __restrict__ gates, const float* __restrict__ bias,
                            float* __restrict__ c, bf16_t* __restrict__ xg, bf16_t* __restrict__ xo) {
  int idx = blockIdx.x * blockDim.x + threadIdx.x;  // 32*512
  if (idx >= BB * HH) return;
  int b = idx >> 9, h = idx & (HH - 1);
  const float* g = gates + (size_t)b * NG;
  float ig = g[h]            + bias[h];
  float fg = g[HH + h]       + bias[HH + h];
  float gg = g[2 * HH + h]   + bias[2 * HH + h];
  float og = g[3 * HH + h]   + bias[3 * HH + h];
  float cn = sigm(fg) * c[idx] + sigm(ig) * tanhf(gg);
  float sn = sigm(og) * tanhf(cn);
  c[idx] = cn;
  bf16_t sb = (bf16_t)sn;
  xg[(size_t)b * KG + HD2 + h] = sb;   // s for next step's [ct | s] and sv-GEMM
  xo[(size_t)b * KG + h] = sb;         // s_new for this step's [s_new | ct]
}

// per-row vocab softmax + write p_gen + accumulate NLL
__global__ void vocab_softmax_kernel(const float* __restrict__ logits, const int* __restrict__ t_idx,
                                     const float* __restrict__ t_mask, float* __restrict__ nll_acc,
                                     float* __restrict__ p_out, int t) {
  const int b = blockIdx.x;          // 32 blocks
  const int tid = threadIdx.x;       // 512 threads
  const float* row = logits + (size_t)b * VV;
  __shared__ float red[512];
  float m = -INFINITY;
  for (int v = tid; v < VV; v += 512) m = fmaxf(m, row[v]);
  red[tid] = m; __syncthreads();
  for (int off = 256; off; off >>= 1) { if (tid < off) red[tid] = fmaxf(red[tid], red[tid + off]); __syncthreads(); }
  m = red[0]; __syncthreads();
  float sum = 0.f;
  for (int v = tid; v < VV; v += 512) sum += expf(row[v] - m);
  red[tid] = sum; __syncthreads();
  for (int off = 256; off; off >>= 1) { if (tid < off) red[tid] += red[tid + off]; __syncthreads(); }
  sum = red[0];
  float inv = 1.f / sum;
  float* p = p_out + ((size_t)t * BB + b) * VV;
  for (int v = tid; v < VV; v += 512) p[v] = expf(row[v] - m) * inv;
  if (tid == 0) {
    int idx = t_idx[t * BB + b];
    nll_acc[b] += (m + logf(sum) - row[idx]) * t_mask[t * BB + b];
  }
}

__global__ void loss_kernel(const float* __restrict__ nll_acc, const float* __restrict__ t_mask,
                            float* __restrict__ out) {
  int b = threadIdx.x;  // 32
  if (b >= BB) return;
  float ms = 0.f;
  for (int t = 0; t < TT; ++t) ms += t_mask[t * BB + b];
  out[b] = nll_acc[b] / ms;
}

// ---------------------------------------------------------------------------
// Host-side launch
// ---------------------------------------------------------------------------
extern "C" void kernel_launch(void* const* d_in, const int* in_sizes, int n_in,
                              void* d_out, int out_size, void* d_ws, size_t ws_size,
                              hipStream_t stream) {
  (void)in_sizes; (void)n_in; (void)out_size; (void)ws_size;
  const float* h_s    = (const float*)d_in[0];
  const float* s_mask = (const float*)d_in[1];
  const float* w_a    = (const float*)d_in[2];
  const float* v_a    = (const float*)d_in[3];
  const float* u_a    = (const float*)d_in[4];
  const float* w_b    = (const float*)d_in[5];
  const float* v_b    = (const float*)d_in[6];
  const float* W_ih   = (const float*)d_in[7];
  const float* W_hh   = (const float*)d_in[8];
  const float* b_ih   = (const float*)d_in[9];
  const float* b_hh   = (const float*)d_in[10];
  const float* t_mask = (const float*)d_in[11];
  const int*   t_idx  = (const int*)d_in[12];
  float* out = (float*)d_out;

  // workspace carve-up (256B aligned)
  char* ws = (char*)d_ws;
  size_t off = 0;
  auto carve = [&](size_t bytes) { char* p = ws + off; off = (off + bytes + 255) & ~(size_t)255; return p; };
  bf16_t* WbT   = (bf16_t*)carve((size_t)VV * KG * sizeof(bf16_t));   // 98.3 MB  (L2-resident across steps)
  bf16_t* WgT   = (bf16_t*)carve((size_t)NG * KG * sizeof(bf16_t));   // 6.3 MB
  bf16_t* vaT   = (bf16_t*)carve((size_t)HH * HH * sizeof(bf16_t));
  bf16_t* waT   = (bf16_t*)carve((size_t)HH * HD2 * sizeof(bf16_t));
  bf16_t* hs_bf = (bf16_t*)carve((size_t)SS * BB * HD2 * sizeof(bf16_t));
  float*  wa_h  = (float*)carve((size_t)SS * BB * HH * sizeof(float));
  float*  sv    = (float*)carve((size_t)BB * HH * sizeof(float));
  float*  scores= (float*)carve((size_t)SS * BB * sizeof(float));
  float*  alpha = (float*)carve((size_t)SS * BB * sizeof(float));
  bf16_t* xg    = (bf16_t*)carve((size_t)BB * KG * sizeof(bf16_t));   // [ct | s]
  bf16_t* xo    = (bf16_t*)carve((size_t)BB * KG * sizeof(bf16_t));   // [s_new | ct]
  float*  gates = (float*)carve((size_t)BB * NG * sizeof(float));
  float*  logits= (float*)carve((size_t)BB * VV * sizeof(float));
  float*  cst   = (float*)carve((size_t)BB * HH * sizeof(float));
  float*  bias  = (float*)carve((size_t)NG * sizeof(float));
  float*  nll   = (float*)carve((size_t)BB * sizeof(float));

  // zero recurrent / accumulated state every call (graph-safe memsets)
  hipMemsetAsync(cst, 0, (size_t)BB * HH * sizeof(float), stream);
  hipMemsetAsync(xg,  0, (size_t)BB * KG * sizeof(bf16_t), stream);
  hipMemsetAsync(xo,  0, (size_t)BB * KG * sizeof(bf16_t), stream);
  hipMemsetAsync(nll, 0, (size_t)BB * sizeof(float), stream);

  // one-time weight conversion (bf16, K-contiguous per output column)
  dim3 tb(32, 8);
  transpose_cvt_kernel<<<dim3(VV / 32, HH / 32), tb, 0, stream>>>(w_b, WbT, HH, VV, KG, 0);
  transpose_cvt_kernel<<<dim3(VV / 32, HD2 / 32), tb, 0, stream>>>(v_b, WbT, HD2, VV, KG, HH);
  transpose_cvt_kernel<<<dim3(HH / 32, HH / 32), tb, 0, stream>>>(v_a, vaT, HH, HH, HH, 0);
  transpose_cvt_kernel<<<dim3(HH / 32, HD2 / 32), tb, 0, stream>>>(w_a, waT, HD2, HH, HD2, 0);
  {
    size_t n = (size_t)NG * KG;
    build_wgt_kernel<<<(unsigned)((n + 255) / 256), 256, 0, stream>>>(W_ih, W_hh, WgT);
  }
  {
    size_t n = (size_t)SS * BB * HD2;
    cvt_bf16_kernel<<<(unsigned)((n + 255) / 256), 256, 0, stream>>>(h_s, hs_bf, n);
  }
  bias_kernel<<<(NG + 255) / 256, 256, 0, stream>>>(b_ih, b_hh, bias);

  // precompute wa_h = h_s @ w_a : M=4096, N=512, K=1024 (WMMA, 2x2 tiles)
  {
    int m_groups = (SS * BB) / 32, n_groups = (HH / 16) / 2;
    int waves = m_groups * n_groups;
    gemm_wmma<2, 2><<<(waves * 32 + 255) / 256, 256, 0, stream>>>(
        hs_bf, HD2, waT, wa_h, HH, m_groups, n_groups, HD2);
  }

  float* p_gen = out + BB;  // output layout: loss[B] then p_gen[T,B,V]

  for (int t = 0; t < TT; ++t) {
    // sv = s @ v_a : M=32, N=512, K=512 (s lives at xg + 1024, row stride KG)
    {
      int n_groups = (HH / 16) / 2;
      gemm_wmma<2, 2><<<(n_groups * 32 + 255) / 256, 256, 0, stream>>>(
          xg + HD2, KG, vaT, sv, HH, 1, n_groups, HH);
    }
    attn_scores_kernel<<<(SS * BB * 32 + 255) / 256, 256, 0, stream>>>(wa_h, sv, u_a, scores);
    alpha_kernel<<<BB, SS, 0, stream>>>(scores, s_mask, alpha);
    context_kernel<<<BB, 256, 0, stream>>>(alpha, h_s, xg, xo);
    // gates = [ct|s] @ [W_ih;W_hh]^T : M=32, N=2048, K=1536
    {
      int n_groups = (NG / 16) / 2;
      gemm_wmma<2, 2><<<(n_groups * 32 + 255) / 256, 256, 0, stream>>>(
          xg, KG, WgT, gates, NG, 1, n_groups, KG);
    }
    lstm_kernel<<<(BB * HH + 255) / 256, 256, 0, stream>>>(gates, bias, cst, xg, xo);
    // logits = [s_new|ct] @ [w_b;v_b] : M=32, N=32000, K=1536 (dominant GEMM)
    {
      int n_groups = (VV / 16) / 2;   // 1000 waves
      gemm_wmma<2, 2><<<(n_groups * 32 + 255) / 256, 256, 0, stream>>>(
          xo, KG, WbT, logits, VV, 1, n_groups, KG);
    }
    vocab_softmax_kernel<<<BB, 512, 0, stream>>>(logits, t_idx, t_mask, nll, p_gen, t);
  }
  loss_kernel<<<1, BB, 0, stream>>>(nll, t_mask, out);
}
